// VQModule_48163763257765
// MI455X (gfx1250) — compile-verified, matrix operand order checked by reading, not run on hardware
//
#include <hip/hip_runtime.h>

#define N_ROWS 16384
#define C_DIM  256
#define KC     8192
#define CB_SCALE 4096.0f

typedef __attribute__((ext_vector_type(16))) _Float16 v16h;
typedef __attribute__((ext_vector_type(8)))  _Float16 v8h;
typedef __attribute__((ext_vector_type(4)))  _Float16 v4h;
typedef __attribute__((ext_vector_type(8)))  float    v8f;
typedef __attribute__((ext_vector_type(4)))  float    v4f;
typedef __attribute__((__vector_size__(4 * sizeof(int)))) int v4i;

#define AS1 __attribute__((address_space(1)))
#define AS3 __attribute__((address_space(3)))

#ifndef __has_builtin
#define __has_builtin(x) 0
#endif
#if __has_builtin(__builtin_amdgcn_global_load_async_to_lds_b128)
#define HAVE_ASYNC_LDS 1
#else
#define HAVE_ASYNC_LDS 0
#endif

static __device__ __forceinline__ void wait_async0() {
#if __has_builtin(__builtin_amdgcn_s_wait_asynccnt)
    __builtin_amdgcn_s_wait_asynccnt(0);
#else
    asm volatile("s_wait_asynccnt 0x0" ::: "memory");
#endif
}

// ---------------- f32 -> f16 conversion (optionally scaled) ----------------
__global__ void cvt_f16_kernel(const float* __restrict__ src,
                               _Float16* __restrict__ dst,
                               int n4, float scale) {
    int i = blockIdx.x * blockDim.x + threadIdx.x;
    if (i < n4) {
        v4f v = ((const v4f*)src)[i];
        v4h h;
        h[0] = (_Float16)(v[0] * scale);
        h[1] = (_Float16)(v[1] * scale);
        h[2] = (_Float16)(v[2] * scale);
        h[3] = (_Float16)(v[3] * scale);
        ((v4h*)dst)[i] = h;
    }
}

// ------- per-code squared norm of the SCALED f16 codebook (f32 accum) ------
__global__ void cnorm_kernel(const _Float16* __restrict__ ch,
                             float* __restrict__ cnorm) {
    int row  = blockIdx.x * 8 + (threadIdx.x >> 5);
    int lane = threadIdx.x & 31;
    const _Float16* p = ch + (size_t)row * C_DIM + lane * 8;
    v8h d = *(const v8h*)p;
    float s = 0.f;
#pragma unroll
    for (int i = 0; i < 8; ++i) { float f = (float)d[i]; s += f * f; }
#pragma unroll
    for (int m = 16; m >= 1; m >>= 1) s += __shfl_xor(s, m, 32);
    if (lane == 0) cnorm[row] = s;
}

// ---------------- main fused GEMM + argmin (WMMA f16->f32) -----------------
static __device__ __forceinline__ v16h make_frag(const _Float16* p) {
    // 16-bit A/B fragment: halves [0..7] at p, halves [8..15] at p+16 elements
    v8h lo = *(const v8h*)p;
    v8h hi = *(const v8h*)(p + 16);
    v16h r;
#pragma unroll
    for (int i = 0; i < 8; ++i) { r[i] = lo[i]; r[i + 8] = hi[i]; }
    return r;
}

#if HAVE_ASYNC_LDS
static __device__ __forceinline__ void stage_async(const _Float16* gsrc,
                                                   _Float16* ldst, int tid) {
    const AS1 char* s = (const AS1 char*)gsrc;
    AS3 char*       d = (AS3 char*)ldst;
#pragma unroll
    for (int t = 0; t < 16; ++t) {
        int off = (tid + 128 * t) * 16;               // 128 thr x 16 x 16B = 32 KB
        __builtin_amdgcn_global_load_async_to_lds_b128(
            (AS1 v4i*)(s + off), (AS3 v4i*)(d + off), 0, 0);
    }
}
#endif

static __device__ __forceinline__ void compute_tile(
        const _Float16* __restrict__ buf, int k0,
        const v16h (&a)[2][8], float (&bestV)[2][8], int (&bestI)[2][8],
        const float* __restrict__ cnorm, int j, int h) {
#pragma unroll
    for (int kt = 0; kt < 4; ++kt) {
        v8f acc0 = {0.f, 0.f, 0.f, 0.f, 0.f, 0.f, 0.f, 0.f};
        v8f acc1 = {0.f, 0.f, 0.f, 0.f, 0.f, 0.f, 0.f, 0.f};
        const _Float16* brow = buf + (kt * 16 + j) * C_DIM;
#pragma unroll
        for (int c = 0; c < 8; ++c) {
            v16h b = make_frag(brow + 32 * c + 8 * h);   // shared B fragment
            acc0 = __builtin_amdgcn_wmma_f32_16x16x32_f16(
                     false, a[0][c], false, b, (short)0, acc0, false, false);
            acc1 = __builtin_amdgcn_wmma_f32_16x16x32_f16(
                     false, a[1][c], false, b, (short)0, acc1, false, false);
        }
        const int   k  = k0 + kt * 16 + j;
        const float cn = cnorm[k];
#pragma unroll
        for (int r = 0; r < 8; ++r) {
            // surrogate distance: s^2*(||c||^2 - 2 z.c), argmin-equivalent
            float d0 = cn - (2.0f * CB_SCALE) * acc0[r];
            float d1 = cn - (2.0f * CB_SCALE) * acc1[r];
            if (d0 < bestV[0][r]) { bestV[0][r] = d0; bestI[0][r] = k; }
            if (d1 < bestV[1][r]) { bestV[1][r] = d1; bestI[1][r] = k; }
        }
    }
}

__launch_bounds__(128)
__global__ void vq_argmin_kernel(const _Float16* __restrict__ xh,
                                 const _Float16* __restrict__ ch,
                                 const float* __restrict__ cnorm,
                                 int* __restrict__ idx_out) {
    __shared__ __align__(16) _Float16 tile[2][64 * C_DIM];   // 2 x 32 KB

    const int tid  = threadIdx.x;
    const int wave = tid >> 5;
    const int lane = tid & 31;
    const int j    = lane & 15;     // column-within-16 / row-within-16
    const int h    = lane >> 4;     // half selector for 16-bit K layout
    const int rowBase = (blockIdx.x * 4 + wave) * 32;        // 32 rows per wave

    // Preload A fragments: two 16-row tiles (32 x 256 f16 -> 128 VGPRs)
    v16h a[2][8];
#pragma unroll
    for (int tdx = 0; tdx < 2; ++tdx) {
        const _Float16* arow = xh + (size_t)(rowBase + 16 * tdx + j) * C_DIM;
#pragma unroll
        for (int c = 0; c < 8; ++c)
            a[tdx][c] = make_frag(arow + 32 * c + 8 * h);
    }

    float bestV[2][8];
    int   bestI[2][8];
#pragma unroll
    for (int tdx = 0; tdx < 2; ++tdx)
#pragma unroll
        for (int r = 0; r < 8; ++r) { bestV[tdx][r] = 3.4e38f; bestI[tdx][r] = 0; }

#if HAVE_ASYNC_LDS
    // ---- double-buffered async global->LDS pipeline, one barrier per tile ----
    stage_async(ch, &tile[0][0], tid);
    int cur = 0;
    for (int k0 = 0; k0 < KC; k0 += 64) {
        wait_async0();          // this thread's writes into tile[cur] complete
        __syncthreads();        // publish tile[cur]; all reads of tile[cur^1] done
        if (k0 + 64 < KC)
            stage_async(ch + (size_t)(k0 + 64) * C_DIM, &tile[cur ^ 1][0], tid);
        compute_tile(&tile[cur][0], k0, a, bestV, bestI, cnorm, j, h);
        cur ^= 1;
    }
#else
    // ---- fallback: VGPR-staged single buffer, two barriers per tile ----
    for (int k0 = 0; k0 < KC; k0 += 64) {
        __syncthreads();
        const v8h* gsrc = (const v8h*)(ch + (size_t)k0 * C_DIM);
        v8h*       ldst = (v8h*)&tile[0][0];
#pragma unroll
        for (int t = 0; t < 32; ++t)
            ldst[tid + 128 * t] = gsrc[tid + 128 * t];
        __syncthreads();
        compute_tile(&tile[0][0], k0, a, bestV, bestI, cnorm, j, h);
    }
#endif

    // Cross-lane argmin over the 16 column-lanes (same h group), tie -> min idx
#pragma unroll
    for (int tdx = 0; tdx < 2; ++tdx) {
#pragma unroll
        for (int r = 0; r < 8; ++r) {
            float bv = bestV[tdx][r];
            int   bi = bestI[tdx][r];
#pragma unroll
            for (int m = 1; m < 16; m <<= 1) {
                float ov = __shfl_xor(bv, m, 32);
                int   oi = __shfl_xor(bi, m, 32);
                if (ov < bv || (ov == bv && oi < bi)) { bv = ov; bi = oi; }
            }
            if (j == 0)                       // C-layout: M = r + 8h
                idx_out[rowBase + 16 * tdx + 8 * h + r] = bi;
        }
    }
}

// -------- gather codebook rows -> qe, per-row squared error, indices -------
__global__ void finalize_kernel(const float* __restrict__ x,
                                const float* __restrict__ cb,
                                const int* __restrict__ idx,
                                float* __restrict__ qe,
                                float* __restrict__ idx_out_f,
                                float* __restrict__ rowsum) {
    int row  = blockIdx.x * 8 + (threadIdx.x >> 5);
    int lane = threadIdx.x & 31;
    int k    = idx[row];
    const float* xr = x  + (size_t)row * C_DIM;
    const float* cr = cb + (size_t)k   * C_DIM;
    float*       qr = qe + (size_t)row * C_DIM;
    float s = 0.f;
#pragma unroll
    for (int t = 0; t < 2; ++t) {
        int c = lane * 4 + t * 128;
        v4f cv = *(const v4f*)(cr + c);
        v4f xv = *(const v4f*)(xr + c);
        *(v4f*)(qr + c) = cv;
#pragma unroll
        for (int i = 0; i < 4; ++i) { float df = xv[i] - cv[i]; s += df * df; }
    }
#pragma unroll
    for (int m = 16; m >= 1; m >>= 1) s += __shfl_xor(s, m, 32);
    if (lane == 0) { rowsum[row] = s; idx_out_f[row] = (float)k; }
}

// --------- deterministic single-block reduction for commit_loss ------------
__global__ void loss_kernel(const float* __restrict__ rowsum,
                            float* __restrict__ out) {
    __shared__ float sm[256];
    float s = 0.f;
    for (int i = threadIdx.x; i < N_ROWS; i += 256) s += rowsum[i];
    sm[threadIdx.x] = s;
    __syncthreads();
    for (int m = 128; m > 0; m >>= 1) {
        if (threadIdx.x < m) sm[threadIdx.x] += sm[threadIdx.x + m];
        __syncthreads();
    }
    if (threadIdx.x == 0) out[0] = sm[0] / (float)(N_ROWS * C_DIM);
}

extern "C" void kernel_launch(void* const* d_in, const int* in_sizes, int n_in,
                              void* d_out, int out_size, void* d_ws, size_t ws_size,
                              hipStream_t stream) {
    const float* x  = (const float*)d_in[0];   // [16384, 256] f32
    const float* cb = (const float*)d_in[1];   // [8192, 256]  f32
    float* out = (float*)d_out;

    // Workspace layout (~12.7 MB)
    _Float16* xh    = (_Float16*)d_ws;                       // 8 MB
    _Float16* ch    = xh + (size_t)N_ROWS * C_DIM;           // 4 MB
    float*    cnorm = (float*)(ch + (size_t)KC * C_DIM);     // 32 KB
    int*      idxws = (int*)(cnorm + KC);                    // 64 KB
    float*    rowsm = (float*)(idxws + N_ROWS);              // 64 KB

    // Output layout: qe | commit_loss | indices (as float)
    float* qe    = out;
    float* lossp = out + (size_t)N_ROWS * C_DIM;
    float* idxf  = lossp + 1;

    int nx4 = N_ROWS * C_DIM / 4;
    int nc4 = KC * C_DIM / 4;
    cvt_f16_kernel<<<(nx4 + 255) / 256, 256, 0, stream>>>(x,  xh, nx4, 1.0f);
    cvt_f16_kernel<<<(nc4 + 255) / 256, 256, 0, stream>>>(cb, ch, nc4, CB_SCALE);
    cnorm_kernel  <<<KC / 8,       256, 0, stream>>>(ch, cnorm);
    vq_argmin_kernel<<<N_ROWS / 128, 128, 0, stream>>>(xh, ch, cnorm, idxws);
    finalize_kernel<<<N_ROWS / 8,  256, 0, stream>>>(x, cb, idxws, qe, idxf, rowsm);
    loss_kernel   <<<1,            256, 0, stream>>>(rowsm, lossp);
}
